// ClsNet_16295105921315
// MI455X (gfx1250) — compile-verified, matrix operand order checked by reading, not run on hardware
//
#include <hip/hip_runtime.h>
#include <hip/hip_bf16.h>
#include <cstdint>

// ---------------- problem constants ----------------
constexpr int Bsz = 8;
constexpr int LX  = 512;
constexpr int LY  = 512;
constexpr int S   = LX + LY + 3;   // 1027
constexpr int Sp  = 1040;          // padded: multiple of 16, 8*Sp multiple of 128
constexpr int D   = 1024;
constexpr int H   = 16;
constexpr int HD  = 64;
constexpr int FF  = 2048;
constexpr int NKV = 2048;          // K and V outputs concatenated
constexpr int NKSTEPS = D / 32;    // 32 K-steps of 32

// ---------------- WMMA / async types ----------------
typedef __attribute__((ext_vector_type(16))) __bf16 v16bf;
typedef __attribute__((ext_vector_type(8)))  float  v8f;
typedef __attribute__((ext_vector_type(4)))  int    v4i;

union Frag16 { v16bf v; uint32_t u[8]; };

// CDNA5 async global->LDS path (ASYNCcnt), guarded so absence degrades to sync staging.
#if defined(__HIP_DEVICE_COMPILE__) && defined(__has_builtin)
#if __has_builtin(__builtin_amdgcn_global_load_async_to_lds_b128) && \
    __has_builtin(__builtin_amdgcn_s_wait_asynccnt)
#define USE_ASYNC_LDS 1
#endif
#endif
#ifndef USE_ASYNC_LDS
#define USE_ASYNC_LDS 0
#endif

#define AS_GLOBAL __attribute__((address_space(1)))
#define AS_LDS    __attribute__((address_space(3)))

#if USE_ASYNC_LDS
static __device__ __forceinline__ void async_b128(const uint32_t* g, uint32_t* l) {
    // builtin expects (global v4i*, lds v4i*, imm offset, imm cpol); go via uintptr_t
    // (int->pointer casts are always legal; generic LDS addrs keep the offset in low 32b)
    __builtin_amdgcn_global_load_async_to_lds_b128(
        (AS_GLOBAL v4i*)(uintptr_t)g,
        (AS_LDS    v4i*)(uintptr_t)l,
        0, 0);
}
#endif

static __device__ __forceinline__ uint16_t f2bf(float f) {
    uint32_t u = __float_as_uint(f);
    u += 0x7FFFu + ((u >> 16) & 1u);       // round-to-nearest-even
    return (uint16_t)(u >> 16);
}

// ---------------- 1) merge x/y/cls/sep into padded bf16 sequence ----------------
__global__ __launch_bounds__(256) void merge_kernel(
    const float* __restrict__ x, const float* __restrict__ y,
    const float* __restrict__ cls, const float* __restrict__ sep,
    const float* __restrict__ px,  const float* __restrict__ py,
    const int* __restrict__ x_len, const int* __restrict__ y_len,
    uint16_t* __restrict__ seq_bf, float* __restrict__ seq0)
{
    const int row = blockIdx.x;            // [0, Bsz*Sp)
    const int b = row / Sp, s = row % Sp;
    const int lx = x_len[b], ly = y_len[b];
    for (int d = threadIdx.x; d < D; d += 256) {
        float v = 0.0f;
        if (s < S) {
            if (s == 0)                              v = cls[d];
            else if (s <= lx)                        v = x[((size_t)b * LX + (s - 1)) * D + d] + px[d];
            else if (s == lx + 1)                    v = sep[d];
            else if (s <= lx + ly + 1)               v = y[((size_t)b * LY + (s - lx - 2)) * D + d] + py[d];
            else if (s == lx + ly + 2)               v = sep[d];
        }
        seq_bf[(size_t)row * D + d] = f2bf(v);
        if (s == 0) seq0[b * D + d] = v;
    }
}

// ---------------- 2) W_kv (rows 1024..3071 of Wqkv) -> bf16 ----------------
__global__ __launch_bounds__(256) void pack_wkv(const float* __restrict__ Wqkv,
                                                uint16_t* __restrict__ wkv_bf)
{
    const int n = blockIdx.x;              // [0, NKV)
    const float* src = Wqkv + (size_t)(D + n) * D;
    for (int k = threadIdx.x; k < D; k += 256)
        wkv_bf[(size_t)n * D + k] = f2bf(src[k]);
}

// ---------------- 3) KV projection: (Bsz*Sp x D) @ (D x NKV) bf16 WMMA ----------------
// LDS tiles are row-major (pitch 20 u32, 16B-aligned rows): fragments load as 2x ds_load_b128.
__global__ __launch_bounds__(256) void kv_gemm(
    const uint32_t* __restrict__ seq_bf,   // [Bsz*Sp][D/2] bf16-pairs
    const uint32_t* __restrict__ wkv_bf,   // [NKV][D/2]    bf16-pairs
    const float* __restrict__ bqkv,
    float* __restrict__ kv)                // [Bsz*Sp][NKV] fp32
{
    __shared__ uint32_t As[2][128][20];    // [buf][m][k-pair], pitch 20 keeps 16B align
    __shared__ uint32_t Bs[2][128][20];    // [buf][n][k-pair]

    const int tid   = threadIdx.x;
    const int lane  = tid & 31;
    const int wave  = tid >> 5;            // 8 waves
    const int wm    = wave & 3;            // 4 waves along M
    const int wn    = wave >> 2;           // 2 waves along N
    const int l15   = lane & 15;
    const int lhalf = lane >> 4;           // 0/1  (K-pair half per ISA 16-bit layout)
    const int m0    = blockIdx.x * 128;
    const int n0    = blockIdx.y * 128;

    // staging decomposition: 512 b128 chunks per matrix per K-step, 2 per thread
    const int c0 = tid, c1 = tid + 256;
    const int r0 = c0 >> 2, q0 = (c0 & 3) * 4;   // row, u32 offset within row
    const int r1 = c1 >> 2, q1 = (c1 & 3) * 4;

    const uint32_t* gA0 = seq_bf + (size_t)(m0 + r0) * (D / 2) + q0;
    const uint32_t* gA1 = seq_bf + (size_t)(m0 + r1) * (D / 2) + q1;
    const uint32_t* gB0 = wkv_bf + (size_t)(n0 + r0) * (D / 2) + q0;
    const uint32_t* gB1 = wkv_bf + (size_t)(n0 + r1) * (D / 2) + q1;

    v8f acc[2][4];
    #pragma unroll
    for (int mi = 0; mi < 2; ++mi)
        #pragma unroll
        for (int ni = 0; ni < 4; ++ni)
            #pragma unroll
            for (int e = 0; e < 8; ++e) acc[mi][ni][e] = 0.0f;

    // fragment build + 8 WMMAs from buffer `buf`
    auto compute = [&](int buf) {
        Frag16 afr[2];
        #pragma unroll
        for (int mi = 0; mi < 2; ++mi) {
            const uint32_t* arow = &As[buf][wm * 32 + mi * 16 + l15][0];
            uint4 lo = *reinterpret_cast<const uint4*>(arow + lhalf * 4);      // K pairs 0..7 / 8..15
            uint4 hi = *reinterpret_cast<const uint4*>(arow + 8 + lhalf * 4);  // K pairs 16..23 / 24..31
            afr[mi].u[0] = lo.x; afr[mi].u[1] = lo.y; afr[mi].u[2] = lo.z; afr[mi].u[3] = lo.w;
            afr[mi].u[4] = hi.x; afr[mi].u[5] = hi.y; afr[mi].u[6] = hi.z; afr[mi].u[7] = hi.w;
        }
        Frag16 bfr[4];
        #pragma unroll
        for (int ni = 0; ni < 4; ++ni) {
            const uint32_t* brow = &Bs[buf][wn * 64 + ni * 16 + l15][0];
            uint4 lo = *reinterpret_cast<const uint4*>(brow + lhalf * 4);
            uint4 hi = *reinterpret_cast<const uint4*>(brow + 8 + lhalf * 4);
            bfr[ni].u[0] = lo.x; bfr[ni].u[1] = lo.y; bfr[ni].u[2] = lo.z; bfr[ni].u[3] = lo.w;
            bfr[ni].u[4] = hi.x; bfr[ni].u[5] = hi.y; bfr[ni].u[6] = hi.z; bfr[ni].u[7] = hi.w;
        }
        #pragma unroll
        for (int mi = 0; mi < 2; ++mi)
            #pragma unroll
            for (int ni = 0; ni < 4; ++ni)
                acc[mi][ni] = __builtin_amdgcn_wmma_f32_16x16x32_bf16(
                    false, afr[mi].v, false, bfr[ni].v,
                    (short)0, acc[mi][ni], false, false);
    };

#if USE_ASYNC_LDS
    // -------- double-buffered async global->LDS pipeline (ASYNCcnt) --------
    auto issue = [&](int kt, int buf) {
        const int ko = kt * 16;  // u32 offset along K
        async_b128(gA0 + ko, &As[buf][r0][q0]);
        async_b128(gA1 + ko, &As[buf][r1][q1]);
        async_b128(gB0 + ko, &Bs[buf][r0][q0]);
        async_b128(gB1 + ko, &Bs[buf][r1][q1]);
    };
    issue(0, 0);
    for (int kt = 0; kt < NKSTEPS; ++kt) {
        const int buf = kt & 1;
        if (kt + 1 < NKSTEPS) {
            issue(kt + 1, buf ^ 1);                 // prefetch next stage
            __builtin_amdgcn_s_wait_asynccnt(4);    // in-order: stage kt has landed
        } else {
            __builtin_amdgcn_s_wait_asynccnt(0);
        }
        __syncthreads();                            // all waves' stage-kt data visible
        compute(buf);
        __syncthreads();                            // buf free for stage kt+2
    }
#else
    // -------- synchronous fallback staging --------
    for (int kt = 0; kt < NKSTEPS; ++kt) {
        const int ko = kt * 16;
        uint4 a0 = *reinterpret_cast<const uint4*>(gA0 + ko);
        uint4 a1 = *reinterpret_cast<const uint4*>(gA1 + ko);
        uint4 b0 = *reinterpret_cast<const uint4*>(gB0 + ko);
        uint4 b1 = *reinterpret_cast<const uint4*>(gB1 + ko);
        __syncthreads();
        *reinterpret_cast<uint4*>(&As[0][r0][q0]) = a0;
        *reinterpret_cast<uint4*>(&As[0][r1][q1]) = a1;
        *reinterpret_cast<uint4*>(&Bs[0][r0][q0]) = b0;
        *reinterpret_cast<uint4*>(&Bs[0][r1][q1]) = b1;
        __syncthreads();
        compute(0);
    }
#endif

    // ---- epilogue: fused bias, fp32 store (KV stays L2-resident for attention) ----
    #pragma unroll
    for (int mi = 0; mi < 2; ++mi)
        #pragma unroll
        for (int ni = 0; ni < 4; ++ni) {
            const int gn = n0 + wn * 64 + ni * 16 + l15;
            const float bias = bqkv[D + gn];
            #pragma unroll
            for (int e = 0; e < 8; ++e) {
                const int gm = m0 + wm * 32 + mi * 16 + lhalf * 8 + e;
                kv[(size_t)gm * NKV + gn] = acc[mi][ni][e] + bias;
            }
        }
}

// ---------------- 4) q at position 0 only ----------------
__global__ __launch_bounds__(256) void q0_gemv(
    const float* __restrict__ seq0, const float* __restrict__ Wqkv,
    const float* __restrict__ bqkv, float* __restrict__ q0)
{
    const int g = blockIdx.x * 256 + threadIdx.x;   // [0, Bsz*D)
    const int b = g / D, d = g % D;
    const float4* s4 = reinterpret_cast<const float4*>(seq0 + (size_t)b * D);
    const float4* w4 = reinterpret_cast<const float4*>(Wqkv + (size_t)d * D);
    float acc = bqkv[d];
    for (int k = 0; k < D / 4; ++k) {
        float4 sv = s4[k], wv = w4[k];
        acc += sv.x * wv.x + sv.y * wv.y + sv.z * wv.z + sv.w * wv.w;
    }
    q0[g] = acc;
}

// ---------------- 5) single-query attention per (b,h) ----------------
__global__ __launch_bounds__(256) void attn_kernel(
    const float* __restrict__ q0, const float* __restrict__ kv,
    const int* __restrict__ x_len, const int* __restrict__ y_len,
    float* __restrict__ o0)
{
    __shared__ float sc[1056];
    __shared__ float qs[HD];
    __shared__ float red[256];
    __shared__ float part[4][HD];

    const int b = blockIdx.x >> 4, h = blockIdx.x & 15;
    const int tid = threadIdx.x;
    const int lim = x_len[b] + y_len[b] + 2;      // inclusive last valid pos

    if (tid < HD) qs[tid] = q0[b * D + h * HD + tid];
    __syncthreads();

    float lmax = -3.4e38f;
    for (int s = tid; s <= lim; s += 256) {
        const float* krow = kv + (size_t)(b * Sp + s) * NKV + h * HD;
        float dot = 0.0f;
        #pragma unroll 8
        for (int j = 0; j < HD; ++j) dot += qs[j] * krow[j];
        dot *= 0.125f;                             // 1/sqrt(64)
        sc[s] = dot;
        lmax = fmaxf(lmax, dot);
    }
    red[tid] = lmax; __syncthreads();
    for (int o = 128; o; o >>= 1) { if (tid < o) red[tid] = fmaxf(red[tid], red[tid + o]); __syncthreads(); }
    const float mx = red[0]; __syncthreads();

    float lsum = 0.0f;
    for (int s = tid; s <= lim; s += 256) { float e = __expf(sc[s] - mx); sc[s] = e; lsum += e; }
    red[tid] = lsum; __syncthreads();
    for (int o = 128; o; o >>= 1) { if (tid < o) red[tid] += red[tid + o]; __syncthreads(); }
    const float inv = 1.0f / red[0];
    __syncthreads();

    const int j = tid & 63, grp = tid >> 6;
    float a = 0.0f;
    for (int s = grp; s <= lim; s += 4)
        a += sc[s] * kv[(size_t)(b * Sp + s) * NKV + D + h * HD + j];
    part[grp][j] = a; __syncthreads();
    if (tid < HD)
        o0[b * D + h * HD + tid] =
            (part[0][tid] + part[1][tid] + part[2][tid] + part[3][tid]) * inv;
}

// ---------------- 6) out-proj + residual + LN1 (one block per batch) ----------------
__global__ __launch_bounds__(1024) void oproj_ln1(
    const float* __restrict__ o0, const float* __restrict__ Wo,
    const float* __restrict__ bo, const float* __restrict__ seq0,
    const float* __restrict__ g, const float* __restrict__ bb,
    float* __restrict__ h0)
{
    __shared__ float so[D];
    __shared__ float red[1024];
    const int b = blockIdx.x, d = threadIdx.x;
    so[d] = o0[b * D + d]; __syncthreads();

    const float4* w4 = reinterpret_cast<const float4*>(Wo + (size_t)d * D);
    const float4* s4 = reinterpret_cast<const float4*>(so);
    float acc = bo[d];
    for (int k = 0; k < D / 4; ++k) {
        float4 sv = s4[k], wv = w4[k];
        acc += sv.x * wv.x + sv.y * wv.y + sv.z * wv.z + sv.w * wv.w;
    }
    const float t = acc + seq0[b * D + d];

    red[d] = t; __syncthreads();
    for (int o = 512; o; o >>= 1) { if (d < o) red[d] += red[d + o]; __syncthreads(); }
    const float mean = red[0] * (1.0f / D); __syncthreads();
    const float c = t - mean;
    red[d] = c * c; __syncthreads();
    for (int o = 512; o; o >>= 1) { if (d < o) red[d] += red[d + o]; __syncthreads(); }
    const float var = red[0] * (1.0f / D);
    h0[b * D + d] = c * rsqrtf(var + 1e-5f) * g[d] + bb[d];
}

// ---------------- 7) FFN layer 1 + ReLU ----------------
__global__ __launch_bounds__(1024) void ffn1(
    const float* __restrict__ h0, const float* __restrict__ W1,
    const float* __restrict__ b1, float* __restrict__ f1)
{
    __shared__ float sh[D];
    const int b = blockIdx.x >> 1, half = blockIdx.x & 1;
    const int j = half * 1024 + threadIdx.x;
    sh[threadIdx.x] = h0[b * D + threadIdx.x]; __syncthreads();

    const float4* w4 = reinterpret_cast<const float4*>(W1 + (size_t)j * D);
    const float4* s4 = reinterpret_cast<const float4*>(sh);
    float acc = b1[j];
    for (int k = 0; k < D / 4; ++k) {
        float4 sv = s4[k], wv = w4[k];
        acc += sv.x * wv.x + sv.y * wv.y + sv.z * wv.z + sv.w * wv.w;
    }
    f1[(size_t)b * FF + j] = fmaxf(acc, 0.0f);
}

// ---------------- 8) FFN layer 2 + residual + LN2 -> out ----------------
__global__ __launch_bounds__(1024) void ffn2_ln2(
    const float* __restrict__ f1, const float* __restrict__ W2,
    const float* __restrict__ b2, const float* __restrict__ h0,
    const float* __restrict__ g, const float* __restrict__ bb,
    float* __restrict__ out)
{
    __shared__ float sf[FF];
    __shared__ float red[1024];
    const int b = blockIdx.x, d = threadIdx.x;
    sf[d]        = f1[(size_t)b * FF + d];
    sf[d + 1024] = f1[(size_t)b * FF + d + 1024];
    __syncthreads();

    const float4* w4 = reinterpret_cast<const float4*>(W2 + (size_t)d * FF);
    const float4* s4 = reinterpret_cast<const float4*>(sf);
    float acc = b2[d];
    for (int k = 0; k < FF / 4; ++k) {
        float4 sv = s4[k], wv = w4[k];
        acc += sv.x * wv.x + sv.y * wv.y + sv.z * wv.z + sv.w * wv.w;
    }
    const float t = acc + h0[b * D + d];

    red[d] = t; __syncthreads();
    for (int o = 512; o; o >>= 1) { if (d < o) red[d] += red[d + o]; __syncthreads(); }
    const float mean = red[0] * (1.0f / D); __syncthreads();
    const float c = t - mean;
    red[d] = c * c; __syncthreads();
    for (int o = 512; o; o >>= 1) { if (d < o) red[d] += red[d + o]; __syncthreads(); }
    const float var = red[0] * (1.0f / D);
    out[b * D + d] = c * rsqrtf(var + 1e-5f) * g[d] + bb[d];
}

// ---------------- host launcher ----------------
extern "C" void kernel_launch(void* const* d_in, const int* in_sizes, int n_in,
                              void* d_out, int out_size, void* d_ws, size_t ws_size,
                              hipStream_t stream) {
    (void)in_sizes; (void)n_in; (void)out_size; (void)ws_size;

    const float* x     = (const float*)d_in[0];
    const float* y     = (const float*)d_in[1];
    const float* cls   = (const float*)d_in[2];
    const float* sep   = (const float*)d_in[3];
    const float* px    = (const float*)d_in[4];
    const float* py    = (const float*)d_in[5];
    const float* Wqkv  = (const float*)d_in[6];
    const float* bqkv  = (const float*)d_in[7];
    const float* Wo    = (const float*)d_in[8];
    const float* bo    = (const float*)d_in[9];
    const float* W1    = (const float*)d_in[10];
    const float* b1    = (const float*)d_in[11];
    const float* W2    = (const float*)d_in[12];
    const float* b2    = (const float*)d_in[13];
    const float* ln1_g = (const float*)d_in[14];
    const float* ln1_b = (const float*)d_in[15];
    const float* ln2_g = (const float*)d_in[16];
    const float* ln2_b = (const float*)d_in[17];
    const int*   x_len = (const int*)d_in[18];
    const int*   y_len = (const int*)d_in[19];
    float* out = (float*)d_out;

    // workspace layout (bytes)
    char* ws = (char*)d_ws;
    size_t off = 0;
    auto alloc = [&](size_t bytes) { void* p = ws + off; off = (off + bytes + 255) & ~size_t(255); return p; };
    uint16_t* seq_bf = (uint16_t*)alloc((size_t)Bsz * Sp * D * 2);   // bf16 sequence
    uint16_t* wkv_bf = (uint16_t*)alloc((size_t)NKV * D * 2);        // bf16 W_kv
    float*    kv     = (float*)   alloc((size_t)Bsz * Sp * NKV * 4); // fp32 K|V
    float*    seq0   = (float*)   alloc((size_t)Bsz * D * 4);
    float*    q0     = (float*)   alloc((size_t)Bsz * D * 4);
    float*    o0     = (float*)   alloc((size_t)Bsz * D * 4);
    float*    h0     = (float*)   alloc((size_t)Bsz * D * 4);
    float*    f1     = (float*)   alloc((size_t)Bsz * FF * 4);

    merge_kernel<<<Bsz * Sp, 256, 0, stream>>>(x, y, cls, sep, px, py, x_len, y_len, seq_bf, seq0);
    pack_wkv<<<NKV, 256, 0, stream>>>(Wqkv, wkv_bf);
    kv_gemm<<<dim3((Bsz * Sp) / 128, NKV / 128), 256, 0, stream>>>(
        (const uint32_t*)seq_bf, (const uint32_t*)wkv_bf, bqkv, kv);
    q0_gemv<<<(Bsz * D) / 256, 256, 0, stream>>>(seq0, Wqkv, bqkv, q0);
    attn_kernel<<<Bsz * H, 256, 0, stream>>>(q0, kv, x_len, y_len, o0);
    oproj_ln1<<<Bsz, 1024, 0, stream>>>(o0, Wo, bo, seq0, ln1_g, ln1_b, h0);
    ffn1<<<Bsz * 2, 1024, 0, stream>>>(h0, W1, b1, f1);
    ffn2_ln2<<<Bsz, 1024, 0, stream>>>(f1, W2, b2, h0, ln2_g, ln2_b, out);
}
